// NunchakuZImageFeedForward_28905129902663
// MI455X (gfx1250) — matrix operand order repaired and checked by reading, not compiled, not run on hardware
//
#include <hip/hip_runtime.h>
#include <hip/hip_bf16.h>

typedef __attribute__((ext_vector_type(16))) _Float16 v16h;
typedef __attribute__((ext_vector_type(8)))  _Float16 v8h;
typedef __attribute__((ext_vector_type(8)))  float    v8f;
typedef __attribute__((ext_vector_type(4)))  float    v4f;
typedef __attribute__((ext_vector_type(4)))  int      v4i;

// ---------------------------------------------------------------------------
// Problem constants (from reference setup_inputs)
// ---------------------------------------------------------------------------
constexpr int S = 4096;   // sequence (M)
constexpr int D = 3072;   // model dim
constexpr int F = 8192;   // ffn dim (per half; w13 is D x 2F)

constexpr int BM = 128;   // block tile M
constexpr int BN = 64;    // block tile N (per half in gemm1)
constexpr int BK = 32;    // K step == WMMA K
constexpr int LDSP = 40;  // padded LDS row stride in halves (80B, 16B-aligned)

static __device__ inline v16h cat16(v8h lo, v8h hi) {
  return __builtin_shufflevector(lo, hi, 0,1,2,3,4,5,6,7,8,9,10,11,12,13,14,15);
}

static __device__ inline v8f wmma_f16(v16h a, v16h b, v8f c) {
  return __builtin_amdgcn_wmma_f32_16x16x32_f16(
      /*neg_a=*/false, a, /*neg_b=*/false, b,
      /*c_mod=*/(short)0, c, /*reuse_a=*/false, /*reuse_b=*/false);
}

// ---------------------------------------------------------------------------
// Async global -> LDS copy (16 bytes per lane), ASYNCcnt-tracked.
// Builtin signature on this toolchain: (int4 AS1*, int4 AS3*, imm, imm cpol).
// ---------------------------------------------------------------------------
typedef __attribute__((address_space(1))) v4i gas_v4i;
typedef __attribute__((address_space(3))) v4i las_v4i;

static __device__ inline void async_cp16(const _Float16* gsrc, _Float16* ldst) {
#if __has_builtin(__builtin_amdgcn_global_load_async_to_lds_b128)
  __builtin_amdgcn_global_load_async_to_lds_b128(
      (gas_v4i*)(gsrc), (las_v4i*)(ldst), /*offset=*/0, /*cpol=*/0);
#else
  unsigned lds_addr = (unsigned)(unsigned long long)ldst;
  asm volatile("global_load_async_to_lds_b128 %0, %1, off"
               : : "v"(lds_addr), "v"(gsrc) : "memory");
#endif
}

template <int N>
static __device__ inline void wait_async() {
#if __has_builtin(__builtin_amdgcn_s_wait_asynccnt)
  __builtin_amdgcn_s_wait_asynccnt((short)N);
#else
  asm volatile("s_wait_asynccnt %0" : : "n"(N) : "memory");
#endif
}

// ---------------------------------------------------------------------------
// fp32 -> fp16 elementwise convert (for x)
// ---------------------------------------------------------------------------
__global__ __launch_bounds__(256) void k_cvt_f16(const float* __restrict__ in,
                                                 _Float16* __restrict__ out,
                                                 long n) {
  long i = ((long)blockIdx.x * 256 + threadIdx.x) * 4;
  if (i + 3 < n) {
    v4f v = *(const v4f*)(in + i);
    out[i + 0] = (_Float16)v.x;
    out[i + 1] = (_Float16)v.y;
    out[i + 2] = (_Float16)v.z;
    out[i + 3] = (_Float16)v.w;
  }
}

// ---------------------------------------------------------------------------
// fp32 [R][C] -> fp16 [C][R]  (tiled transpose-convert for w13, w2)
// ---------------------------------------------------------------------------
__global__ __launch_bounds__(256) void k_transpose_cvt(const float* __restrict__ in,
                                                       _Float16* __restrict__ out,
                                                       int R, int C) {
  __shared__ float tile[32][33];
  const int bx = blockIdx.x * 32;
  const int by = blockIdx.y * 32;
  const int tx = threadIdx.x & 31;
  const int ty = threadIdx.x >> 5;
#pragma unroll
  for (int i = 0; i < 32; i += 8)
    tile[ty + i][tx] = in[(long)(by + ty + i) * C + bx + tx];
  __syncthreads();
#pragma unroll
  for (int i = 0; i < 32; i += 8)
    out[(long)(bx + ty + i) * R + by + tx] = (_Float16)tile[tx][ty + i];
}

// ---------------------------------------------------------------------------
// GEMM1 + SwiGLU, async double-buffered, no-unroll steady-state loop.
// ---------------------------------------------------------------------------
__global__ __launch_bounds__(256) void k_gemm1_swiglu(
    const _Float16* __restrict__ xh,    // [S][D]
    const _Float16* __restrict__ w13t,  // [2F][D]
    _Float16* __restrict__ g) {         // [S][F]
  __shared__ __align__(16) _Float16 As[2][BM * LDSP];
  __shared__ __align__(16) _Float16 Bs[2][2][BN * LDSP];

  const int m0   = blockIdx.x * BM;
  const int n0   = blockIdx.y * BN;
  const int tid  = threadIdx.x;
  const int lane = tid & 31;
  const int wave = tid >> 5;
  const int wm   = (wave & 3) * 32;
  const int wn   = (wave >> 2) * 32;
  const int lrow = lane & 15;
  const int lhi  = lane >> 4;

  const int ar0 = tid >> 2,         as0 = tid & 3;
  const int ar1 = (tid + 256) >> 2, as1 = tid & 3;
  const int br  = (tid & 255) >> 2, bs = tid & 3;

  v8f acc[2][2][2] = {};  // [half][mtile][ntile]

  auto issue_tile = [&](int buf, int k0) {
    async_cp16(xh + (long)(m0 + ar0) * D + k0 + as0 * 8,
               &As[buf][ar0 * LDSP + as0 * 8]);
    async_cp16(xh + (long)(m0 + ar1) * D + k0 + as1 * 8,
               &As[buf][ar1 * LDSP + as1 * 8]);
    async_cp16(w13t + (long)(n0 + br) * D + k0 + bs * 8,
               &Bs[buf][0][br * LDSP + bs * 8]);
    async_cp16(w13t + (long)(F + n0 + br) * D + k0 + bs * 8,
               &Bs[buf][1][br * LDSP + bs * 8]);
  };

  auto compute_tile = [&](int cur) {
    v16h afrag[2];
#pragma unroll
    for (int mt = 0; mt < 2; ++mt) {
      const _Float16* base = &As[cur][(wm + mt * 16 + lrow) * LDSP];
      v8h lo = *(const v8h*)(base + lhi * 8);
      v8h hi = *(const v8h*)(base + 16 + lhi * 8);
      afrag[mt] = cat16(lo, hi);
    }
#pragma unroll
    for (int h = 0; h < 2; ++h) {
#pragma unroll
      for (int nt = 0; nt < 2; ++nt) {
        const _Float16* base = &Bs[cur][h][(wn + nt * 16 + lrow) * LDSP + lhi * 8];
        v8h lo = *(const v8h*)(base);
        v8h hi = *(const v8h*)(base + 16);
        v16h bfrag = cat16(lo, hi);
        acc[h][0][nt] = wmma_f16(afrag[0], bfrag, acc[h][0][nt]);
        acc[h][1][nt] = wmma_f16(afrag[1], bfrag, acc[h][1][nt]);
      }
    }
  };

  constexpr int NKT = D / BK;  // 96
  issue_tile(0, 0);

#pragma clang loop unroll(disable)
  for (int kt = 0; kt < NKT - 1; ++kt) {
    issue_tile((kt + 1) & 1, (kt + 1) * BK);
    wait_async<4>();   // tile kt landed (4 in-flight belong to tile kt+1)
    __syncthreads();
    compute_tile(kt & 1);
    __syncthreads();   // safe to refill this buffer next round
  }
  // peeled last tile
  wait_async<0>();
  __syncthreads();
  compute_tile((NKT - 1) & 1);

  // epilogue: SwiGLU gate, clamp, f16 store
#pragma unroll
  for (int mt = 0; mt < 2; ++mt) {
#pragma unroll
    for (int nt = 0; nt < 2; ++nt) {
#pragma unroll
      for (int r = 0; r < 8; ++r) {
        float x3 = acc[0][mt][nt][r];
        float x1 = acc[1][mt][nt][r];
        float sl = x1 * (1.0f / (1.0f + __expf(-x1)));
        float v  = sl * x3;
        v = fminf(fmaxf(v, -65504.0f), 65504.0f);
        const int m = m0 + wm + mt * 16 + lhi * 8 + r;
        const int n = n0 + wn + nt * 16 + lrow;
        g[(long)m * F + n] = (_Float16)v;
      }
    }
  }
}

// ---------------------------------------------------------------------------
// GEMM2: out = g @ w2 (via w2t [D][F]), async double-buffered, f32 out.
// ---------------------------------------------------------------------------
__global__ __launch_bounds__(256) void k_gemm2(
    const _Float16* __restrict__ g,     // [S][F]
    const _Float16* __restrict__ w2t,   // [D][F]
    float* __restrict__ out) {          // [S][D]
  __shared__ __align__(16) _Float16 As[2][BM * LDSP];
  __shared__ __align__(16) _Float16 Bs[2][BN * LDSP];

  const int m0   = blockIdx.x * BM;
  const int n0   = blockIdx.y * BN;
  const int tid  = threadIdx.x;
  const int lane = tid & 31;
  const int wave = tid >> 5;
  const int wm   = (wave & 3) * 32;
  const int wn   = (wave >> 2) * 32;
  const int lrow = lane & 15;
  const int lhi  = lane >> 4;

  const int ar0 = tid >> 2,         as0 = tid & 3;
  const int ar1 = (tid + 256) >> 2, as1 = tid & 3;
  const int br  = tid >> 2,         bs = tid & 3;

  v8f acc[2][2] = {};  // [mtile][ntile]

  auto issue_tile = [&](int buf, int k0) {
    async_cp16(g + (long)(m0 + ar0) * F + k0 + as0 * 8,
               &As[buf][ar0 * LDSP + as0 * 8]);
    async_cp16(g + (long)(m0 + ar1) * F + k0 + as1 * 8,
               &As[buf][ar1 * LDSP + as1 * 8]);
    async_cp16(w2t + (long)(n0 + br) * F + k0 + bs * 8,
               &Bs[buf][br * LDSP + bs * 8]);
  };

  auto compute_tile = [&](int cur) {
    v16h afrag[2];
#pragma unroll
    for (int mt = 0; mt < 2; ++mt) {
      const _Float16* base = &As[cur][(wm + mt * 16 + lrow) * LDSP];
      v8h lo = *(const v8h*)(base + lhi * 8);
      v8h hi = *(const v8h*)(base + 16 + lhi * 8);
      afrag[mt] = cat16(lo, hi);
    }
#pragma unroll
    for (int nt = 0; nt < 2; ++nt) {
      const _Float16* base = &Bs[cur][(wn + nt * 16 + lrow) * LDSP + lhi * 8];
      v8h lo = *(const v8h*)(base);
      v8h hi = *(const v8h*)(base + 16);
      v16h bfrag = cat16(lo, hi);
      acc[0][nt] = wmma_f16(afrag[0], bfrag, acc[0][nt]);
      acc[1][nt] = wmma_f16(afrag[1], bfrag, acc[1][nt]);
    }
  };

  constexpr int NKT = F / BK;  // 256
  issue_tile(0, 0);

#pragma clang loop unroll(disable)
  for (int kt = 0; kt < NKT - 1; ++kt) {
    issue_tile((kt + 1) & 1, (kt + 1) * BK);
    wait_async<3>();
    __syncthreads();
    compute_tile(kt & 1);
    __syncthreads();
  }
  wait_async<0>();
  __syncthreads();
  compute_tile((NKT - 1) & 1);

#pragma unroll
  for (int mt = 0; mt < 2; ++mt) {
#pragma unroll
    for (int nt = 0; nt < 2; ++nt) {
#pragma unroll
      for (int r = 0; r < 8; ++r) {
        const int m = m0 + wm + mt * 16 + lhi * 8 + r;
        const int n = n0 + wn + nt * 16 + lrow;
        out[(long)m * D + n] = acc[mt][nt][r];
      }
    }
  }
}

// ---------------------------------------------------------------------------
// Host launcher
// ---------------------------------------------------------------------------
extern "C" void kernel_launch(void* const* d_in, const int* in_sizes, int n_in,
                              void* d_out, int out_size, void* d_ws, size_t ws_size,
                              hipStream_t stream) {
  (void)in_sizes; (void)n_in; (void)out_size; (void)ws_size;

  const float* x   = (const float*)d_in[0];   // [S][D]
  const float* w13 = (const float*)d_in[1];   // [D][2F]
  const float* w2  = (const float*)d_in[2];   // [F][D]
  float* out = (float*)d_out;                 // [S][D]

  char* ws = (char*)d_ws;
  _Float16* xh   = (_Float16*)(ws);
  _Float16* w13t = (_Float16*)(ws + (long)S * D * 2);
  _Float16* w2t  = (_Float16*)(ws + (long)S * D * 2 + (long)2 * F * D * 2);
  _Float16* gbuf = (_Float16*)(ws + (long)S * D * 2 + (long)2 * F * D * 2
                                  + (long)D * F * 2);

  {
    long n = (long)S * D;
    k_cvt_f16<<<(int)(n / 4 / 256), 256, 0, stream>>>(x, xh, n);
  }
  {
    dim3 grid(2 * F / 32, D / 32);
    k_transpose_cvt<<<grid, 256, 0, stream>>>(w13, w13t, D, 2 * F);
  }
  {
    dim3 grid(D / 32, F / 32);
    k_transpose_cvt<<<grid, 256, 0, stream>>>(w2, w2t, F, D);
  }
  {
    dim3 grid(S / BM, F / BN);
    k_gemm1_swiglu<<<grid, 256, 0, stream>>>(xh, w13t, gbuf);
  }
  {
    dim3 grid(S / BM, D / BN);
    k_gemm2<<<grid, 256, 0, stream>>>(gbuf, w2t, out);
  }
}
